// RelativeCrossAttentionLayer_54537494725149
// MI455X (gfx1250) — compile-verified, matrix operand order checked by reading, not run on hardware
//
#include <hip/hip_runtime.h>
#include <hip/hip_bf16.h>

#define H    8
#define E    512
#define D    64
#define LQ   2048
#define LK   2048
#define BB   2
#define ROWS (LQ * BB)     // 4096
#define EPSV 1e-5f

typedef __attribute__((ext_vector_type(16))) _Float16 v16h;
typedef __attribute__((ext_vector_type(8)))  float    v8f;

__device__ __forceinline__ v8f wmma16(v16h a, v16h b, v8f c) {
  // D = A(16x32 f16) * B(32x16 f16) + C(16x16 f32)
  return __builtin_amdgcn_wmma_f32_16x16x32_f16(false, a, false, b, (short)0, c,
                                                false, false);
}

// ---- CDNA5 async global->LDS staging (ASYNCcnt-tracked) --------------------
__device__ __forceinline__ unsigned lds_off_of(const void* p) {
  // generic pointer to __shared__: low 32 bits are the LDS byte offset
  return (unsigned)(size_t)p;
}

__device__ __forceinline__ void async_copy16(unsigned lds_off, const void* g) {
  // per-lane 16B copy, tracked with ASYNCcnt (ISA 15.18.3 opcode 98)
  asm volatile("global_load_async_to_lds_b128 %0, %1, off"
               :: "v"(lds_off), "v"((unsigned long long)(size_t)g)
               : "memory");
}

template <int N>
__device__ __forceinline__ void wait_async() {
  asm volatile("s_wait_asynccnt %0" :: "n"(N) : "memory");
}

// Load one 16x32 f16 WMMA operand from row-major f32 memory (convert on load).
// Lane layout per CDNA5 ISA: lanes 0-15 -> major r, K = {k0..k0+7, k0+16..k0+23};
// lanes 16-31 -> major r, K = {k0+8..k0+15, k0+24..k0+31}.
__device__ __forceinline__ v16h load_tile_f32(const float* __restrict__ base,
                                              int stride, int major0, int k0) {
  const int lane = threadIdx.x & 31;
  const float* p = base + (size_t)(major0 + (lane & 15)) * stride + k0 +
                   ((lane >> 4) << 3);
  v16h x;
#pragma unroll
  for (int i = 0; i < 8; ++i) x[i] = (_Float16)p[i];
#pragma unroll
  for (int i = 0; i < 8; ++i) x[8 + i] = (_Float16)p[16 + i];
  return x;
}

__device__ __forceinline__ v16h load_tile_f16(const _Float16* __restrict__ base,
                                              int stride, int major0, int k0) {
  const int lane = threadIdx.x & 31;
  const _Float16* p = base + (size_t)(major0 + (lane & 15)) * stride + k0 +
                      ((lane >> 4) << 3);
  v16h x;
#pragma unroll
  for (int i = 0; i < 8; ++i) x[i] = p[i];
#pragma unroll
  for (int i = 0; i < 8; ++i) x[8 + i] = p[16 + i];
  return x;
}

// ---------------------------------------------------------------------------
// Kernel 1: q/k/v = X @ W^T + b, fused rotary (+ q scale), store f16.
// grid = (ROWS/16, E/16, 3), block = 32 (one wave)
// ---------------------------------------------------------------------------
__global__ void k_inproj_rotary(const float* __restrict__ query,
                                const float* __restrict__ value,
                                const float* __restrict__ qpos,
                                const float* __restrict__ vpos,
                                const float* __restrict__ w,
                                const float* __restrict__ bias,
                                _Float16* __restrict__ qh,
                                _Float16* __restrict__ kh,
                                _Float16* __restrict__ vT) {
  const int lane = threadIdx.x & 31;
  const int mt   = blockIdx.x * 16;
  const int nt   = blockIdx.y * 16;
  const int proj = blockIdx.z;  // 0=q 1=k 2=v
  const float* X = (proj == 0) ? query : value;
  const float* W = w + (size_t)proj * E * E;

  v8f acc = {};
#pragma unroll 4
  for (int kk = 0; kk < E; kk += 32) {
    v16h a = load_tile_f32(X, E, mt, kk);
    v16h b = load_tile_f32(W, E, nt, kk);
    acc = wmma16(a, b, acc);
  }

  const int   col  = nt + (lane & 15);
  const float bval = bias[proj * E + col];
  const float* pos = (proj == 0) ? qpos : vpos;
  const int   hidx = col >> 6;
  const int   dd   = col & (D - 1);

#pragma unroll
  for (int r = 0; r < 8; ++r) {
    const int row = mt + r + ((lane >> 4) << 3);
    const int lq  = row / BB;
    const int bb  = row % BB;
    float y = acc[r] + bval;
    float pr = __shfl_xor(y, 1, 32);  // rotary partner (col^1 == lane^1)
    if (proj < 2) {
      const size_t pidx = (((size_t)bb * LQ + lq) * E + col) * 2;
      const float c = pos[pidx], s = pos[pidx + 1];
      const float xr = (col & 1) ? pr : -pr;
      y = y * c + xr * s;
    }
    if (proj == 0) {
      y *= 0.125f;  // d^-1/2, d = 64
      qh[(((size_t)bb * H + hidx) * LQ + lq) * D + dd] = (_Float16)y;
    } else if (proj == 1) {
      kh[(((size_t)bb * H + hidx) * LK + lq) * D + dd] = (_Float16)y;
    } else {
      vT[(((size_t)bb * H + hidx) * D + dd) * LK + lq] = (_Float16)y;
    }
  }
}

// ---------------------------------------------------------------------------
// Kernel 2: per (b,h,q) running softmax max/sum over all k.
// 4 waves/block share each 16x64 k-tile, double-buffered async global->LDS.
// grid = (LQ/64, BB*H), block = 128
// ---------------------------------------------------------------------------
__global__ void k_softmax_stats(const _Float16* __restrict__ qh,
                                const _Float16* __restrict__ kh,
                                float* __restrict__ mmax,
                                float* __restrict__ ssum) {
  __shared__ _Float16 lk[2][16 * D];  // 2 x 2KB k tiles
  const int tid  = threadIdx.x;
  const int lane = tid & 31;
  const int wave = tid >> 5;
  const int qt   = blockIdx.x * 64 + wave * 16;
  const int bh   = blockIdx.y;
  const _Float16* qb = qh + (size_t)bh * LQ * D;
  const _Float16* kb = kh + (size_t)bh * LK * D;
  const unsigned lkbase = lds_off_of(&lk[0][0]);

  const v16h qa0 = load_tile_f16(qb, D, qt, 0);
  const v16h qa1 = load_tile_f16(qb, D, qt, 32);

  float m8[8], s8[8];
#pragma unroll
  for (int r = 0; r < 8; ++r) { m8[r] = -1e30f; s8[r] = 0.f; }

  // stage tile 0 (2KB = 128 threads x 16B, k rows are contiguous in kh)
  async_copy16(lkbase + (unsigned)tid * 16u, kb + (size_t)tid * 8);

  for (int kt = 0; kt < LK; kt += 16) {
    const int cur = (kt >> 4) & 1;
    if (kt + 16 < LK) {
      async_copy16(lkbase + (unsigned)(cur ^ 1) * 2048u + (unsigned)tid * 16u,
                   kb + (size_t)(kt + 16) * D + (size_t)tid * 8);
      wait_async<1>();  // oldest (current tile) complete
    } else {
      wait_async<0>();
    }
    __syncthreads();

    const _Float16* lkc = &lk[cur][0];
    v8f sc = {};
    sc = wmma16(qa0, load_tile_f16(lkc, D, 0, 0), sc);
    sc = wmma16(qa1, load_tile_f16(lkc, D, 0, 32), sc);
#pragma unroll
    for (int r = 0; r < 8; ++r) {
      const float x  = sc[r];
      const float mn = fmaxf(m8[r], x);
      s8[r] = s8[r] * __expf(m8[r] - mn) + __expf(x - mn);
      m8[r] = mn;
    }
    __syncthreads();  // all waves done reading before this buffer is restaged
  }

  // combine across the 16 column-lanes of each half-wave
#pragma unroll
  for (int off = 1; off <= 8; off <<= 1) {
#pragma unroll
    for (int r = 0; r < 8; ++r) {
      const float mo = __shfl_xor(m8[r], off, 32);
      const float so = __shfl_xor(s8[r], off, 32);
      const float mn = fmaxf(m8[r], mo);
      s8[r] = s8[r] * __expf(m8[r] - mn) + so * __expf(mo - mn);
      m8[r] = mn;
    }
  }
  if ((lane & 15) == 0) {
#pragma unroll
    for (int r = 0; r < 8; ++r) {
      const int row = qt + r + ((lane >> 4) << 3);
      mmax[(size_t)bh * LQ + row] = m8[r];
      ssum[(size_t)bh * LQ + row] = s8[r];
    }
  }
}

// ---------------------------------------------------------------------------
// Kernel 3: head-mean attention weights -> d_out part 2 (written once).
// grid = (LQ/16, LK/16, BB), block = 32
// ---------------------------------------------------------------------------
__global__ void k_wmean(const _Float16* __restrict__ qh,
                        const _Float16* __restrict__ kh,
                        const float* __restrict__ mmax,
                        const float* __restrict__ ssum,
                        float* __restrict__ wmean) {
  const int lane = threadIdx.x & 31;
  const int qt = blockIdx.x * 16;
  const int kt = blockIdx.y * 16;
  const int b  = blockIdx.z;
  v8f acc = {};
  for (int h = 0; h < H; ++h) {
    const int bh = b * H + h;
    const _Float16* qb = qh + (size_t)bh * LQ * D;
    const _Float16* kb = kh + (size_t)bh * LK * D;
    v8f sc = {};
    sc = wmma16(load_tile_f16(qb, D, qt, 0),  load_tile_f16(kb, D, kt, 0),  sc);
    sc = wmma16(load_tile_f16(qb, D, qt, 32), load_tile_f16(kb, D, kt, 32), sc);
#pragma unroll
    for (int r = 0; r < 8; ++r) {
      const int row = qt + r + ((lane >> 4) << 3);
      const float m = mmax[(size_t)bh * LQ + row];
      const float s = ssum[(size_t)bh * LQ + row];
      acc[r] += __expf(sc[r] - m) / s;
    }
  }
#pragma unroll
  for (int r = 0; r < 8; ++r) {
    const int row = qt + r + ((lane >> 4) << 3);
    const int col = kt + (lane & 15);
    wmean[((size_t)b * LQ + row) * LK + col] = acc[r] * (1.f / H);
  }
}

// ---------------------------------------------------------------------------
// Kernel 4: attn = softmax(scores) @ v.  4 waves/block (64 q rows) share
// 32-wide k and v chunks staged into LDS with double-buffered async copies;
// per-wave w tile goes through LDS for the A-operand layout.
// grid = (LQ/64, BB*H), block = 128
// ---------------------------------------------------------------------------
__global__ void k_attn(const _Float16* __restrict__ qh,
                       const _Float16* __restrict__ kh,
                       const _Float16* __restrict__ vT,
                       const float* __restrict__ mmax,
                       const float* __restrict__ ssum,
                       _Float16* __restrict__ attn) {
  // per buffer: k chunk 32x64 (4KB) then v chunk 64x32 (4KB)
  __shared__ _Float16 lkv[2][4096];
  __shared__ _Float16 lw[4 * 16 * 32];  // per-wave 16x32 w tile
  const int tid  = threadIdx.x;
  const int lane = tid & 31;
  const int wave = tid >> 5;
  const int qt   = blockIdx.x * 64 + wave * 16;
  const int bh   = blockIdx.y;
  const int b    = bh / H;
  const int h    = bh % H;
  const _Float16* qb = qh + (size_t)bh * LQ * D;
  const _Float16* kb = kh + (size_t)bh * LK * D;
  const _Float16* vb = vT + (size_t)bh * D * LK;
  const unsigned kvbase = lds_off_of(&lkv[0][0]);

  const v16h qa0 = load_tile_f16(qb, D, qt, 0);
  const v16h qa1 = load_tile_f16(qb, D, qt, 32);

  float m8[8], rinv[8];
#pragma unroll
  for (int r = 0; r < 8; ++r) {
    const int row = qt + r + ((lane >> 4) << 3);
    m8[r]   = mmax[(size_t)bh * LQ + row];
    rinv[r] = 1.0f / ssum[(size_t)bh * LQ + row];
  }

  // stage one 8KB chunk: 512 x 16B copies, 4 per thread (2 k + 2 v)
  auto stage = [&](int buf, int kt) {
    const unsigned b0 = kvbase + (unsigned)buf * 8192u;
#pragma unroll
    for (int i = 0; i < 2; ++i) {          // k rows kt..kt+31 are contiguous
      const int c = tid + i * 128;         // 0..255
      async_copy16(b0 + (unsigned)c * 16u, kb + (size_t)kt * D + (size_t)c * 8);
    }
#pragma unroll
    for (int i = 0; i < 2; ++i) {          // v: 64 rows x 64B, stride LK halfs
      const int j = tid + i * 128;         // 0..255
      const int row = j >> 2, seg = j & 3;
      async_copy16(b0 + 4096u + (unsigned)j * 16u,
                   vb + (size_t)row * LK + kt + seg * 8);
    }
  };

  v8f acc[4] = {};
  stage(0, 0);
  for (int kt = 0; kt < LK; kt += 32) {
    const int cur = (kt >> 5) & 1;
    if (kt + 32 < LK) {
      stage(cur ^ 1, kt + 32);
      wait_async<4>();  // the 4 copies of the current chunk are complete
    } else {
      wait_async<0>();
    }
    __syncthreads();

    const _Float16* lkc = &lkv[cur][0];
    const _Float16* lvc = &lkv[cur][2048];
#pragma unroll
    for (int t = 0; t < 2; ++t) {
      v8f sc = {};
      sc = wmma16(qa0, load_tile_f16(lkc, D, 16 * t, 0),  sc);
      sc = wmma16(qa1, load_tile_f16(lkc, D, 16 * t, 32), sc);
#pragma unroll
      for (int r = 0; r < 8; ++r) {
        const int rr = r + ((lane >> 4) << 3);
        const float wv = __expf(sc[r] - m8[r]) * rinv[r];
        lw[wave * 512 + rr * 32 + 16 * t + (lane & 15)] = (_Float16)wv;
      }
    }
    __syncthreads();
    const v16h wa = load_tile_f16(&lw[wave * 512], 32, 0, 0);  // A layout
#pragma unroll
    for (int t = 0; t < 4; ++t)
      acc[t] = wmma16(wa, load_tile_f16(lvc, 32, 16 * t, 0), acc[t]);
    __syncthreads();  // chunk fully consumed before it is restaged
  }

#pragma unroll
  for (int t = 0; t < 4; ++t) {
#pragma unroll
    for (int r = 0; r < 8; ++r) {
      const int q   = qt + r + ((lane >> 4) << 3);
      const int col = h * D + 16 * t + (lane & 15);
      attn[((size_t)q * BB + b) * E + col] = (_Float16)acc[t][r];
    }
  }
}

// ---------------------------------------------------------------------------
// Kernel 5: out_proj GEMM + bias + residual + LayerNorm -> d_out part 1.
// grid = (ROWS/16), block = 256 (8 waves; wave w owns 64 output columns)
// ---------------------------------------------------------------------------
__global__ void k_outproj_ln(const _Float16* __restrict__ attn,
                             const float* __restrict__ ow,
                             const float* __restrict__ ob,
                             const float* __restrict__ lnw,
                             const float* __restrict__ lnb,
                             const float* __restrict__ query,
                             float* __restrict__ out) {
  __shared__ float rsum[16];
  __shared__ float rsq[16];
  const int lane = threadIdx.x & 31;
  const int wave = threadIdx.x >> 5;
  const int mt   = blockIdx.x * 16;
  if (threadIdx.x < 16) { rsum[threadIdx.x] = 0.f; rsq[threadIdx.x] = 0.f; }
  __syncthreads();

  const int n0 = wave * 64;
  v8f acc[4] = {};
  for (int kk = 0; kk < E; kk += 32) {
    const v16h a = load_tile_f16(attn, E, mt, kk);
#pragma unroll
    for (int t = 0; t < 4; ++t)
      acc[t] = wmma16(a, load_tile_f32(ow, E, n0 + 16 * t, kk), acc[t]);
  }

  float lsum[8] = {}, lsq[8] = {};
#pragma unroll
  for (int t = 0; t < 4; ++t) {
    const int col = n0 + 16 * t + (lane & 15);
    const float bv = ob[col];
#pragma unroll
    for (int r = 0; r < 8; ++r) {
      const int row = mt + r + ((lane >> 4) << 3);
      const float x = acc[t][r] + bv + query[(size_t)row * E + col];
      acc[t][r] = x;
      lsum[r] += x;
      lsq[r]  += x * x;
    }
  }
#pragma unroll
  for (int off = 1; off <= 8; off <<= 1) {
#pragma unroll
    for (int r = 0; r < 8; ++r) {
      lsum[r] += __shfl_xor(lsum[r], off, 32);
      lsq[r]  += __shfl_xor(lsq[r], off, 32);
    }
  }
  if ((lane & 15) == 0) {
#pragma unroll
    for (int r = 0; r < 8; ++r) {
      const int rl = r + ((lane >> 4) << 3);
      atomicAdd(&rsum[rl], lsum[r]);  // ds_add_f32
      atomicAdd(&rsq[rl],  lsq[r]);
    }
  }
  __syncthreads();

#pragma unroll
  for (int t = 0; t < 4; ++t) {
    const int col = n0 + 16 * t + (lane & 15);
    const float g  = lnw[col];
    const float b2 = lnb[col];
#pragma unroll
    for (int r = 0; r < 8; ++r) {
      const int rl  = r + ((lane >> 4) << 3);
      const int row = mt + rl;
      const float mu  = rsum[rl] * (1.f / E);
      const float var = rsq[rl] * (1.f / E) - mu * mu;
      const float x = (acc[t][r] - mu) * rsqrtf(var + EPSV);
      out[(size_t)row * E + col] = x * g + b2;
    }
  }
}

// ---------------------------------------------------------------------------
extern "C" void kernel_launch(void* const* d_in, const int* in_sizes, int n_in,
                              void* d_out, int out_size, void* d_ws, size_t ws_size,
                              hipStream_t stream) {
  (void)in_sizes; (void)n_in; (void)out_size; (void)ws_size;
  const float* query = (const float*)d_in[0];
  const float* value = (const float*)d_in[1];
  const float* qpos  = (const float*)d_in[2];
  const float* vpos  = (const float*)d_in[3];
  const float* ipw   = (const float*)d_in[4];
  const float* ipb   = (const float*)d_in[5];
  const float* opw   = (const float*)d_in[6];
  const float* opb   = (const float*)d_in[7];
  const float* lnw   = (const float*)d_in[8];
  const float* lnb   = (const float*)d_in[9];

  char* ws = (char*)d_ws;
  const size_t qkv_half = (size_t)BB * H * LQ * D * sizeof(_Float16);  // 4 MB
  _Float16* qh   = (_Float16*)(ws);
  _Float16* kh   = (_Float16*)(ws + qkv_half);
  _Float16* vT   = (_Float16*)(ws + 2 * qkv_half);
  _Float16* attn = (_Float16*)(ws + 3 * qkv_half);
  float*    mmax = (float*)(ws + 4 * qkv_half);
  float*    ssum = (float*)(ws + 4 * qkv_half + (size_t)BB * H * LQ * sizeof(float));

  float* out_main = (float*)d_out;                       // (Lq, B, E)
  float* out_wm   = out_main + (size_t)LQ * BB * E;      // (B, Lq, Lk)

  dim3 g1(ROWS / 16, E / 16, 3);
  k_inproj_rotary<<<g1, 32, 0, stream>>>(query, value, qpos, vpos, ipw, ipb,
                                         qh, kh, vT);
  dim3 g2(LQ / 64, BB * H);
  k_softmax_stats<<<g2, 128, 0, stream>>>(qh, kh, mmax, ssum);
  dim3 g3(LQ / 16, LK / 16, BB);
  k_wmean<<<g3, 32, 0, stream>>>(qh, kh, mmax, ssum, out_wm);
  dim3 g4(LQ / 64, BB * H);
  k_attn<<<g4, 128, 0, stream>>>(qh, kh, vT, mmax, ssum, attn);
  k_outproj_ln<<<ROWS / 16, 256, 0, stream>>>(attn, opw, opb, lnw, lnb, query,
                                              out_main);
}
